// LightGCN_Symptom_21930103013811
// MI455X (gfx1250) — matrix operand order repaired and checked by reading, not compiled, not run on hardware
//
#include <hip/hip_runtime.h>
#include <hip/hip_bf16.h>
#include <cstdint>

// Problem constants (from reference)
#define U_CNT   100000
#define I_CNT   50000
#define N_CNT   150000
#define D_CNT   64
#define L_CNT   3
#define NNZ_CNT 2400000

typedef __attribute__((ext_vector_type(16))) _Float16 v16h;
typedef __attribute__((ext_vector_type(8)))  float    v8f;

// ---------------------------------------------------------------------------
// WMMA helpers (gfx1250: V_WMMA_F32_16X16X32_F16, wave32)
// ---------------------------------------------------------------------------
static __device__ __forceinline__ v8f wmma_16x16x32(v16h a, v16h b, v8f c) {
  // (neg_a, A, neg_b, B, c_mod, C, reuse_a, reuse_b)
  return __builtin_amdgcn_wmma_f32_16x16x32_f16(false, a, false, b, (short)0, c,
                                                false, false);
}

// Same-wave LDS producer->consumer fence (DS ops are in-order per wave on
// CDNA5; we only need to stop compiler reordering and wait the DS counter).
static __device__ __forceinline__ void wave_lds_fence() {
  __builtin_amdgcn_wave_barrier();
  asm volatile("s_wait_dscnt 0" ::: "memory");
  __builtin_amdgcn_wave_barrier();
}

// A-fragment (16x32, f16) of row-major f32 matrix X[ld], tile origin (row0,k0).
// ISA layout: lanes 0-15 / 16-31 both hold M=lane&15; lane-half hi selects the
// K interleave: VGPR j, elem e -> K = (j<4?0:16) + hi*8 + (j&3)*2 + e.
static __device__ __forceinline__ v16h load_a_f32(const float* __restrict__ X,
                                                  int row0, int k0, int ld) {
  const int lane = threadIdx.x & 31;
  const int m = lane & 15, hi = lane >> 4;
  const float* p = X + (size_t)(row0 + m) * ld + k0;
  v16h a;
#pragma unroll
  for (int j = 0; j < 8; ++j) {
    const int kb = ((j < 4) ? 0 : 16) + hi * 8 + (j & 3) * 2;
    const float2 f = *(const float2*)(p + kb);   // K-pairs are contiguous
    a[2 * j]     = (_Float16)f.x;
    a[2 * j + 1] = (_Float16)f.y;
  }
  return a;
}

// Same A layout but sourced from an f16 LDS tile [16 rows x 32 K], stride 32.
static __device__ __forceinline__ v16h load_a_lds(const _Float16* __restrict__ h) {
  const int lane = threadIdx.x & 31;
  const int m = lane & 15, hi = lane >> 4;
  const _Float16* p = h + m * 32;
  v16h a;
#pragma unroll
  for (int j = 0; j < 8; ++j) {
    const int kb = ((j < 4) ? 0 : 16) + hi * 8 + (j & 3) * 2;
    a[2 * j]     = p[kb];
    a[2 * j + 1] = p[kb + 1];
  }
  return a;
}

// B-fragment from the pre-packed layout: tile t stores lane l's v16h at
// P + t*512 + l*16  (32 contiguous bytes per lane -> two b128 loads).
static __device__ __forceinline__ v16h load_b_packed(const _Float16* __restrict__ P,
                                                     int tile) {
  const int lane = threadIdx.x & 31;
  return *(const v16h*)(P + (size_t)tile * 512 + lane * 16);
}

// ---------------------------------------------------------------------------
// Weight pre-pack: f32 row-major W[K,cols] -> per-tile WMMA B-fragment f16.
// Tile (kt,nt) covers rows kt*32..+31, cols nt*16..+15; tile id = kt*ntN + nt.
// ISA B layout: lane n=lane&15 holds column nt*16+n; VGPR j holds the K pair
// K = (j<4?0:16) + hi*8 + (j&3)*2 + {0,1} (same interleave as A).
// One wave packs one tile.
// ---------------------------------------------------------------------------
__global__ __launch_bounds__(256) void pack_b_kernel(const float* __restrict__ W,
                                                     int ld, int ntN,
                                                     _Float16* __restrict__ out,
                                                     int ntiles) {
  const int wid = (int)((blockIdx.x * (size_t)blockDim.x + threadIdx.x) >> 5);
  const int lane = threadIdx.x & 31;
  if (wid >= ntiles) return;
  const int kt = wid / ntN, nt = wid % ntN;
  const int n = lane & 15, hi = lane >> 4;
  const float* p = W + (size_t)(kt * 32 + hi * 8) * ld + nt * 16 + n;
  v16h b;
#pragma unroll
  for (int j = 0; j < 8; ++j) {
    const int kb = ((j < 4) ? 0 : 16) + (j & 3) * 2;
    b[2 * j]     = (_Float16)p[kb * ld];
    b[2 * j + 1] = (_Float16)p[(kb + 1) * ld];
  }
  *(v16h*)(out + (size_t)wid * 512 + lane * 16) = b;
}

// ---------------------------------------------------------------------------
// Fused 2-layer MLP: Y = relu(X @ W1 + b1) @ W2 + b2
// X:[M,64] W1:[64,512] W2:[512,64] (both pre-packed). One wave = 16 rows; the
// 16x512 hidden tile is streamed 32 units at a time through a 1KB per-wave
// LDS staging buffer (C-layout -> A-layout cross-lane shuffle).
// W1p tiles: kt in {0,1}, ntN=32 -> tile = kt*32 + hb.
// W2p tiles: kt in 0..15,  ntN=4 -> tile = kb*4 + nb.
// ---------------------------------------------------------------------------
#define NW 8  // waves per 256-thread block

__global__ __launch_bounds__(256) void mlp2_kernel(
    const float* __restrict__ X, const _Float16* __restrict__ W1p,
    const float* __restrict__ b1, const _Float16* __restrict__ W2p,
    const float* __restrict__ b2, float* __restrict__ Y, int M) {
  __shared__ _Float16 hstage[NW][16 * 32];
  const int wave = threadIdx.x >> 5, lane = threadIdx.x & 31;
  const int row0 = (blockIdx.x * NW + wave) * 16;
  if (row0 >= M) return;  // M is a multiple of 16; no partial tiles
  const int n = lane & 15, hi = lane >> 4;

  const v16h a0 = load_a_f32(X, row0, 0, 64);
  const v16h a1 = load_a_f32(X, row0, 32, 64);

  v8f yacc[4];
#pragma unroll
  for (int nb = 0; nb < 4; ++nb) {
    const float bv = b2[nb * 16 + n];
#pragma unroll
    for (int r = 0; r < 8; ++r) yacc[nb][r] = bv;
  }

  _Float16* hl = hstage[wave];
  for (int kb = 0; kb < 16; ++kb) {  // 32 hidden units per iteration
#pragma unroll
    for (int sub = 0; sub < 2; ++sub) {
      const int hb = kb * 2 + sub;     // hidden 16-col block index (0..31)
      v8f h;
      const float bv = b1[hb * 16 + n];
#pragma unroll
      for (int r = 0; r < 8; ++r) h[r] = bv;
      h = wmma_16x16x32(a0, load_b_packed(W1p, hb), h);
      h = wmma_16x16x32(a1, load_b_packed(W1p, 32 + hb), h);
      // ReLU + stage in C-layout: lane holds (row = r + hi*8, col = n)
#pragma unroll
      for (int r = 0; r < 8; ++r) {
        float v = h[r];
        v = v > 0.f ? v : 0.f;
        hl[(r + hi * 8) * 32 + sub * 16 + n] = (_Float16)v;
      }
    }
    wave_lds_fence();
    const v16h ah = load_a_lds(hl);
#pragma unroll
    for (int nb = 0; nb < 4; ++nb)
      yacc[nb] = wmma_16x16x32(ah, load_b_packed(W2p, kb * 4 + nb), yacc[nb]);
    wave_lds_fence();  // before the next iteration overwrites the stage
  }

#pragma unroll
  for (int nb = 0; nb < 4; ++nb)
#pragma unroll
    for (int r = 0; r < 8; ++r)
      Y[(size_t)(row0 + r + hi * 8) * 64 + nb * 16 + n] = yacc[nb][r];
}

// ---------------------------------------------------------------------------
// Sigmoid-gate combine:  g = sigmoid([A0|A1] @ W + b);  out = g*mixA+(1-g)*mixB
// [M,128] @ [128,64] with pre-packed W (kt 0..3, ntN=4 -> tile = k*4 + nb).
// One wave = 16 rows, 16 WMMAs per wave.
// ---------------------------------------------------------------------------
__global__ __launch_bounds__(256) void gate_combine_kernel(
    const float* __restrict__ A0, const float* __restrict__ A1,
    const _Float16* __restrict__ Wp, const float* __restrict__ b,
    const float* __restrict__ mixA, const float* __restrict__ mixB,
    float* __restrict__ out, int M) {
  const int wave = threadIdx.x >> 5, lane = threadIdx.x & 31;
  const int row0 = (blockIdx.x * NW + wave) * 16;
  if (row0 >= M) return;
  const int n = lane & 15, hi = lane >> 4;

  v16h af[4];
  af[0] = load_a_f32(A0, row0, 0, 64);
  af[1] = load_a_f32(A0, row0, 32, 64);
  af[2] = load_a_f32(A1, row0, 0, 64);
  af[3] = load_a_f32(A1, row0, 32, 64);

#pragma unroll
  for (int nb = 0; nb < 4; ++nb) {
    v8f c;
    const float bv = b[nb * 16 + n];
#pragma unroll
    for (int r = 0; r < 8; ++r) c[r] = bv;
#pragma unroll
    for (int k = 0; k < 4; ++k)
      c = wmma_16x16x32(af[k], load_b_packed(Wp, k * 4 + nb), c);
#pragma unroll
    for (int r = 0; r < 8; ++r) {
      const size_t idx = (size_t)(row0 + r + hi * 8) * 64 + nb * 16 + n;
      const float g = 1.f / (1.f + __expf(-c[r]));
      out[idx] = g * mixA[idx] + (1.f - g) * mixB[idx];
    }
  }
}

// ---------------------------------------------------------------------------
// Tiny gate MLP (128 -> 32 -> 1, sigmoid) + mix, one wave per row.
// lane j owns hidden unit j; wave butterfly-reduce for the scalar output.
// Optionally accumulates the mixed row into `acc`.
// ---------------------------------------------------------------------------
__global__ __launch_bounds__(256) void gate_mix_kernel(
    const float* __restrict__ catA, const float* __restrict__ catB,
    const float* __restrict__ gW1, const float* __restrict__ gb1,
    const float* __restrict__ gW2, const float* __restrict__ gb2,
    const float* __restrict__ mixA, const float* __restrict__ mixB,
    float* __restrict__ out, float* __restrict__ acc, int rows) {
  const int wid = (int)((blockIdx.x * (size_t)blockDim.x + threadIdx.x) >> 5);
  const int lane = threadIdx.x & 31;
  if (wid >= rows) return;
  const float* xa = catA + (size_t)wid * 64;
  const float* xb = catB + (size_t)wid * 64;

  float h = gb1[lane];
#pragma unroll 4
  for (int k = 0; k < 64; ++k) h += xa[k] * gW1[k * 32 + lane];
#pragma unroll 4
  for (int k = 0; k < 64; ++k) h += xb[k] * gW1[(64 + k) * 32 + lane];
  h = h > 0.f ? h : 0.f;

  float part = h * gW2[lane];
#pragma unroll
  for (int off = 16; off > 0; off >>= 1) part += __shfl_xor(part, off, 32);
  const float g = 1.f / (1.f + __expf(-(part + gb2[0])));

  const float* ma = mixA + (size_t)wid * 64;
  const float* mb = mixB + (size_t)wid * 64;
  float* po = out + (size_t)wid * 64;
  float* pa = acc ? acc + (size_t)wid * 64 : nullptr;
#pragma unroll
  for (int c = lane; c < 64; c += 32) {
    const float v = g * ma[c] + (1.f - g) * mb[c];
    po[c] = v;
    if (pa) pa[c] += v;
  }
}

// ---------------------------------------------------------------------------
// COO SpMM scatter: out[row] += val * x[col].
// 16 lanes x 4 dims per edge (float4 gather, 4 f32 atomics); 2 edges per wave.
// x (38 MB) lives in L2; atomics resolve at L2.
// ---------------------------------------------------------------------------
__global__ __launch_bounds__(256) void spmm_scatter_kernel(
    const int* __restrict__ row, const int* __restrict__ col,
    const float* __restrict__ val, const float* __restrict__ x,
    float* __restrict__ out, int nnz) {
  const long long t = blockIdx.x * (long long)blockDim.x + threadIdx.x;
  const int e = (int)(t >> 4);
  if (e >= nnz) return;
  const int d = ((int)t & 15) * 4;
  const int r = row[e], c = col[e];
  const float v = val[e];
  const float4 xx = *(const float4*)(x + (size_t)c * 64 + d);
  float* o = out + (size_t)r * 64 + d;
  atomicAdd(o + 0, v * xx.x);
  atomicAdd(o + 1, v * xx.y);
  atomicAdd(o + 2, v * xx.z);
  atomicAdd(o + 3, v * xx.w);
}

// ---------------------------------------------------------------------------
// Elementwise utilities (float4; all element counts are multiples of 4)
// ---------------------------------------------------------------------------
__global__ __launch_bounds__(256) void zero4_kernel(float4* __restrict__ p, long long n4) {
  const long long i = blockIdx.x * (long long)blockDim.x + threadIdx.x;
  if (i < n4) p[i] = make_float4(0.f, 0.f, 0.f, 0.f);
}
__global__ __launch_bounds__(256) void copy4_kernel(const float4* __restrict__ s,
                                                    float4* __restrict__ d, long long n4) {
  const long long i = blockIdx.x * (long long)blockDim.x + threadIdx.x;
  if (i < n4) d[i] = s[i];
}
__global__ __launch_bounds__(256) void copy_add4_kernel(const float4* __restrict__ s,
                                                        float4* __restrict__ d,
                                                        float4* __restrict__ a,
                                                        long long n4) {
  const long long i = blockIdx.x * (long long)blockDim.x + threadIdx.x;
  if (i < n4) {
    const float4 v = s[i];
    d[i] = v;
    const float4 av = a[i];
    a[i] = make_float4(av.x + v.x, av.y + v.y, av.z + v.z, av.w + v.w);
  }
}
__global__ __launch_bounds__(256) void scale4_kernel(const float4* __restrict__ s,
                                                     float4* __restrict__ d, long long n4,
                                                     float f) {
  const long long i = blockIdx.x * (long long)blockDim.x + threadIdx.x;
  if (i < n4) {
    const float4 v = s[i];
    d[i] = make_float4(v.x * f, v.y * f, v.z * f, v.w * f);
  }
}

static inline unsigned cdiv(long long a, long long b) { return (unsigned)((a + b - 1) / b); }

// ---------------------------------------------------------------------------
extern "C" void kernel_launch(void* const* d_in, const int* in_sizes, int n_in,
                              void* d_out, int out_size, void* d_ws, size_t ws_size,
                              hipStream_t stream) {
  const float* emb_user    = (const float*)d_in[0];
  const float* emb_item    = (const float*)d_in[1];
  const float* symptom_emb = (const float*)d_in[2];
  const float* herb_emb    = (const float*)d_in[3];
  const float* sp_W1 = (const float*)d_in[4];
  const float* sp_b1 = (const float*)d_in[5];
  const float* sp_W2 = (const float*)d_in[6];
  const float* sp_b2 = (const float*)d_in[7];
  const float* hp_W1 = (const float*)d_in[8];
  const float* hp_b1 = (const float*)d_in[9];
  const float* hp_W2 = (const float*)d_in[10];
  const float* hp_b2 = (const float*)d_in[11];
  const float* sg_W  = (const float*)d_in[12];
  const float* sg_b  = (const float*)d_in[13];
  const float* hg_W  = (const float*)d_in[14];
  const float* hg_b  = (const float*)d_in[15];
  const float* gate_W1 = (const float*)d_in[16];  // (4,128,32)
  const float* gate_b1 = (const float*)d_in[17];  // (4,32)
  const float* gate_W2 = (const float*)d_in[18];  // (4,32,1)
  const float* gate_b2 = (const float*)d_in[19];  // (4,1)
  const float* base_val = (const float*)d_in[20];
  const float* co_val   = (const float*)d_in[21];
  const int* base_row = (const int*)d_in[22];
  const int* base_col = (const int*)d_in[23];
  const int* co_row   = (const int*)d_in[24];
  const int* co_col   = (const int*)d_in[25];

  const size_t NE = (size_t)N_CNT * 64;  // 9.6M floats per node buffer
  float* ws      = (float*)d_ws;
  float* all_emb = ws;              // [N,64]
  float* acc     = ws + NE;         // [N,64]
  float* base    = ws + 2 * NE;     // [N,64]
  float* co      = ws + 3 * NE;     // [N,64]
  float* proj_s  = base;                         // parked in `base` pre-layers
  float* proj_h  = base + (size_t)U_CNT * 64;

  // Packed f16 weight fragments (after the 4 node buffers)
  _Float16* pk = (_Float16*)(ws + 4 * NE);
  _Float16* sp_W1p = pk;               // 64 tiles  * 512 halfs
  _Float16* sp_W2p = pk + 32768;       // 64 tiles
  _Float16* hp_W1p = pk + 65536;       // 64 tiles
  _Float16* hp_W2p = pk + 98304;       // 64 tiles
  _Float16* sg_Wp  = pk + 131072;      // 16 tiles
  _Float16* hg_Wp  = pk + 139264;      // 16 tiles

  // 0) pre-pack weights into WMMA B-fragment layout (runs every call)
  pack_b_kernel<<<cdiv(64, 8), 256, 0, stream>>>(sp_W1, 512, 32, sp_W1p, 64);
  pack_b_kernel<<<cdiv(64, 8), 256, 0, stream>>>(sp_W2, 64, 4, sp_W2p, 64);
  pack_b_kernel<<<cdiv(64, 8), 256, 0, stream>>>(hp_W1, 512, 32, hp_W1p, 64);
  pack_b_kernel<<<cdiv(64, 8), 256, 0, stream>>>(hp_W2, 64, 4, hp_W2p, 64);
  pack_b_kernel<<<cdiv(16, 8), 256, 0, stream>>>(sg_W, 64, 4, sg_Wp, 16);
  pack_b_kernel<<<cdiv(16, 8), 256, 0, stream>>>(hg_W, 64, 4, hg_Wp, 16);

  // 1) projection MLPs (WMMA, fused)
  mlp2_kernel<<<cdiv(U_CNT / 16, NW), 256, 0, stream>>>(symptom_emb, sp_W1p, sp_b1,
                                                        sp_W2p, sp_b2, proj_s, U_CNT);
  mlp2_kernel<<<cdiv(I_CNT / 16, NW), 256, 0, stream>>>(herb_emb, hp_W1p, hp_b1,
                                                        hp_W2p, hp_b2, proj_h, I_CNT);

  // 2) sigmoid-gate combine (WMMA 128->64)
  gate_combine_kernel<<<cdiv(U_CNT / 16, NW), 256, 0, stream>>>(
      emb_user, proj_s, sg_Wp, sg_b, emb_user, proj_s, all_emb, U_CNT);
  gate_combine_kernel<<<cdiv(I_CNT / 16, NW), 256, 0, stream>>>(
      emb_item, proj_h, hg_Wp, hg_b, emb_item, proj_h, all_emb + (size_t)U_CNT * 64,
      I_CNT);

  // 3) layer-0 gate mixes (in place; each row fully read before written)
  gate_mix_kernel<<<cdiv(U_CNT, 8), 256, 0, stream>>>(
      all_emb, symptom_emb, gate_W1, gate_b1, gate_W2, gate_b2, all_emb, symptom_emb,
      all_emb, nullptr, U_CNT);
  gate_mix_kernel<<<cdiv(I_CNT, 8), 256, 0, stream>>>(
      all_emb + (size_t)U_CNT * 64, herb_emb, gate_W1, gate_b1, gate_W2, gate_b2,
      all_emb + (size_t)U_CNT * 64, herb_emb, all_emb + (size_t)U_CNT * 64, nullptr,
      I_CNT);

  // 4) acc = all_emb
  copy4_kernel<<<cdiv(NE / 4, 256), 256, 0, stream>>>((const float4*)all_emb,
                                                      (float4*)acc, (long long)(NE / 4));

  // 5) propagation layers
  for (int layer = 1; layer <= L_CNT; ++layer) {
    const float* gW1 = gate_W1 + (size_t)layer * 128 * 32;
    const float* gb1 = gate_b1 + (size_t)layer * 32;
    const float* gW2 = gate_W2 + (size_t)layer * 32;
    const float* gb2 = gate_b2 + (size_t)layer;

    zero4_kernel<<<cdiv(NE / 2, 256), 256, 0, stream>>>((float4*)base,
                                                        (long long)(NE / 2));
    spmm_scatter_kernel<<<cdiv((long long)NNZ_CNT * 16, 256), 256, 0, stream>>>(
        base_row, base_col, base_val, all_emb, base, NNZ_CNT);
    spmm_scatter_kernel<<<cdiv((long long)NNZ_CNT * 16, 256), 256, 0, stream>>>(
        co_row, co_col, co_val, all_emb, co, NNZ_CNT);

    // users: all_emb[:U] = base[:U]; acc[:U] += base[:U]
    copy_add4_kernel<<<cdiv((long long)U_CNT * 16, 256), 256, 0, stream>>>(
        (const float4*)base, (float4*)all_emb, (float4*)acc,
        (long long)U_CNT * 16);
    // items: g = gate([base_items|herb]); fused = g*base_items+(1-g)*co_items
    gate_mix_kernel<<<cdiv(I_CNT, 8), 256, 0, stream>>>(
        base + (size_t)U_CNT * 64, herb_emb, gW1, gb1, gW2, gb2,
        base + (size_t)U_CNT * 64, co + (size_t)U_CNT * 64,
        all_emb + (size_t)U_CNT * 64, acc + (size_t)U_CNT * 64, I_CNT);
  }

  // 6) out = acc / (L+1)
  scale4_kernel<<<cdiv(NE / 4, 256), 256, 0, stream>>>(
      (const float4*)acc, (float4*)d_out, (long long)(NE / 4),
      1.f / (float)(L_CNT + 1));
}
